// GraphConvolution_48438641164784
// MI455X (gfx1250) — matrix operand order repaired and checked by reading, not compiled, not run on hardware
//
#include <hip/hip_runtime.h>
#include <hip/hip_bf16.h>

// ---------------------------------------------------------------------------
// GraphConvolution with label-routed experts, tuned for MI455X (gfx1250):
//   support = input @ W[labels]        (tiny: 0.27 GFLOP, VALU f32)
//   out     = adj @ support + bias     (17.2 GFLOP, 256 MB adj stream)
// Big GEMM runs on v_wmma_f32_16x16x32_f16 with f32 accumulate; adj/support
// converted to f16 in registers (inputs are O(1), no range risk).
// Roofline: adj must stream once from HBM (256 MB @ 23.3 TB/s ~ 11 us), so
// the GEMM needs >= 1.5 PFLOP/s effective -> 16-bit WMMA path, not f32 WMMA.
// Only ~30% of f16-WMMA peak is needed to sit on the HBM roofline, so we
// keep the compiler's low-pressure schedule (no spills) rather than forcing
// deeper WMMA pipelining at the cost of scratch traffic.
// ---------------------------------------------------------------------------

typedef __attribute__((ext_vector_type(16))) _Float16 v16h;
typedef __attribute__((ext_vector_type(8)))  float    v8f;

union HFrag {
    v16h     v;
    _Float16 h[16];
    uint4    q[2];
};

constexpr int N_NODES = 8192;
constexpr int IN_F    = 128;
constexpr int OUT_F   = 128;

// ---------------------------------------------------------------------------
// Kernel 1: per-node expert projection.
//   supportT[o][n] = sum_i input[n][i] * W[labels[n]][i][o], stored as f16,
//   TRANSPOSED (o-major, pitch N_NODES) so the GEMM's B-fragments read 16
//   K values per lane as two contiguous runs of 8 with two b128 loads.
// Block: 128 threads handles 16 nodes. Thread t owns output column t.
// ---------------------------------------------------------------------------
__global__ void expert_proj_kernel(const float* __restrict__ input,
                                   const long long* __restrict__ labels,
                                   const float* __restrict__ W,
                                   unsigned short* __restrict__ supportT) {
    __shared__ float s_in[16][IN_F];
    __shared__ int   s_lab[16];

    const int tid = threadIdx.x;          // 0..127 == output feature
    const int n0  = blockIdx.x * 16;

    for (int idx = tid; idx < 16 * IN_F; idx += 128)
        s_in[idx >> 7][idx & 127] = input[(size_t)(n0 + (idx >> 7)) * IN_F + (idx & 127)];
    if (tid < 16) s_lab[tid] = (int)labels[n0 + tid];
    __syncthreads();

    union { _Float16 h[16]; uint4 q[2]; } pk;

    #pragma unroll
    for (int j = 0; j < 16; ++j) {
        const float* w = W + (size_t)s_lab[j] * (IN_F * OUT_F) + tid;
        float acc = 0.f;
        #pragma unroll 8
        for (int i = 0; i < IN_F; ++i)
            acc = fmaf(s_in[j][i], w[(size_t)i * OUT_F], acc);
        pk.h[j] = (_Float16)acc;
    }

    // supportT[tid][n0 .. n0+15] : 32 contiguous bytes, two b128 stores.
    uint4* dst = (uint4*)(supportT + (size_t)tid * N_NODES + n0);
    dst[0] = pk.q[0];
    dst[1] = pk.q[1];
}

// ---------------------------------------------------------------------------
// Kernel 2: out[8192][128] = adj[8192][8192] @ support[8192][128] + bias.
// Block = 64 threads (2 waves), M-tile = 32 rows -> 256 blocks.
// Each wave computes a 16x128 strip: 8 accumulator tiles (v8f each).
// supportT K-chunks (128 n x 128 k, f16) staged in LDS (padded pitch).
// A fragment (16x32 of adj) loaded f32 from HBM, converted to f16 in regs
// following the ISA 16-bit A layout:
//   lane<16  : row m0+lane,    K {0..7, 16..23}
//   lane>=16 : row m0+lane-16, K {8..15, 24..31}
// B fragments per the ISA 16-bit B layout:
//   lane<16  : N = nt*16+lane,    K kk..kk+7, kk+16..kk+23
//   lane>=16 : N = nt*16+lane-16, K kk+8..kk+15, kk+24..kk+31
// ---------------------------------------------------------------------------
__global__ void adj_gemm_kernel(const float* __restrict__ adj,
                                const unsigned short* __restrict__ supportT,
                                const float* __restrict__ bias,
                                float* __restrict__ out) {
    constexpr int KC      = 128;   // K elements staged per chunk
    constexpr int PITCH_Q = 17;    // uint4 per n-row (256B data + 16B pad)
    __shared__ uint4 s_b[OUT_F * PITCH_Q];   // 34,816 B

    const int tid  = threadIdx.x;     // 0..63
    const int lane = tid & 31;
    const int wave = tid >> 5;
    const int l15  = lane & 15;
    const int hi   = lane >> 4;       // 0: lanes 0-15, 1: lanes 16-31
    const int m0   = blockIdx.x * 32 + wave * 16;

    v8f acc[8] = {};

    float bias_v[8];
    #pragma unroll
    for (int nt = 0; nt < 8; ++nt) bias_v[nt] = bias[nt * 16 + l15];

    const float* a_row = adj + (size_t)(m0 + l15) * N_NODES + hi * 8;
    // Per-lane base into the LDS B chunk: row (l15 of each 16-wide N tile),
    // K half selected by hi. Quad index; nt advances by 16*PITCH_Q quads.
    const int b_base_q = l15 * PITCH_Q + hi;       // hi*8 elems = 1 quad

    HFrag a[2];   // double-buffered A fragment (no WAR hazard on reuse)

    for (int kc = 0; kc < N_NODES; kc += KC) {
        __syncthreads();
        // Cooperative stage: supportT[:, kc:kc+KC] -> LDS (2048 uint4).
        for (int idx = tid; idx < OUT_F * (KC / 8); idx += 64) {
            const int n = idx >> 4;
            const int q = idx & 15;
            s_b[n * PITCH_Q + q] =
                *(const uint4*)(supportT + (size_t)n * N_NODES + kc + q * 8);
        }
        __syncthreads();

        #pragma unroll
        for (int kk = 0; kk < KC; kk += 32) {
            HFrag& af = a[(kk >> 5) & 1];
            // ---- A fragment: 16x32 tile of adj, f32 -> f16 in regs ----
            const float* ap = a_row + kc + kk;
            const float4 f0 = *(const float4*)(ap + 0);
            const float4 f1 = *(const float4*)(ap + 4);
            const float4 f2 = *(const float4*)(ap + 16);
            const float4 f3 = *(const float4*)(ap + 20);
            af.h[0]  = (_Float16)f0.x; af.h[1]  = (_Float16)f0.y;
            af.h[2]  = (_Float16)f0.z; af.h[3]  = (_Float16)f0.w;
            af.h[4]  = (_Float16)f1.x; af.h[5]  = (_Float16)f1.y;
            af.h[6]  = (_Float16)f1.z; af.h[7]  = (_Float16)f1.w;
            af.h[8]  = (_Float16)f2.x; af.h[9]  = (_Float16)f2.y;
            af.h[10] = (_Float16)f2.z; af.h[11] = (_Float16)f2.w;
            af.h[12] = (_Float16)f3.x; af.h[13] = (_Float16)f3.y;
            af.h[14] = (_Float16)f3.z; af.h[15] = (_Float16)f3.w;

            // ---- 8 WMMAs sharing the A fragment; B from LDS per tile ----
            const int kq = kk / 8;            // quad offset for this k-step
            #pragma unroll
            for (int nt = 0; nt < 8; ++nt) {
                const uint4* bp = &s_b[b_base_q + nt * 16 * PITCH_Q + kq];
                HFrag b;
                b.q[0] = bp[0];
                b.q[1] = bp[2];
                acc[nt] = __builtin_amdgcn_wmma_f32_16x16x32_f16(
                    /*neg_a=*/false, af.v, /*neg_b=*/false, b.v,
                    /*c_mod=*/(short)0, acc[nt],
                    /*reuse_a=*/false, /*reuse_b=*/false);
            }
        }
    }

    // Epilogue: C/D layout -> VGPR r holds M = r + 8*hi, N = l15.
    #pragma unroll
    for (int nt = 0; nt < 8; ++nt) {
        union { v8f v; float f[8]; } d;
        d.v = acc[nt];
        #pragma unroll
        for (int r = 0; r < 8; ++r) {
            const int m = m0 + r + 8 * hi;
            out[(size_t)m * OUT_F + nt * 16 + l15] = d.f[r] + bias_v[nt];
        }
    }
}

// ---------------------------------------------------------------------------
extern "C" void kernel_launch(void* const* d_in, const int* in_sizes, int n_in,
                              void* d_out, int out_size, void* d_ws, size_t ws_size,
                              hipStream_t stream) {
    const float*     input  = (const float*)d_in[0];
    const float*     adj    = (const float*)d_in[1];
    const long long* labels = (const long long*)d_in[2];
    const float*     W      = (const float*)d_in[3];
    const float*     bias   = (const float*)d_in[4];
    float* out = (float*)d_out;

    // Workspace: supportT f16 [OUT_F][N_NODES] = 2 MB
    unsigned short* supportT = (unsigned short*)d_ws;

    expert_proj_kernel<<<N_NODES / 16, 128, 0, stream>>>(input, labels, W, supportT);
    adj_gemm_kernel<<<N_NODES / 32, 64, 0, stream>>>(adj, supportT, bias, out);
}
